// LSTM_85375359910419
// MI455X (gfx1250) — matrix operand order repaired and checked by reading, not compiled, not run on hardware
//
#include <hip/hip_runtime.h>

// ---------------------------------------------------------------------------
// LSTM for MI455X (gfx1250, wave32, WMMA).
// T=512, B=64, E=H=1024.  f16 multiplicands, f32 accumulation via
// v_wmma_f32_16x16x32_f16.  x-projections precomputed in one parallel GEMM;
// recurrence runs as 512 stream-ordered small kernels with fused 4-gate
// tiles + in-register cell update.
// ---------------------------------------------------------------------------

typedef __attribute__((ext_vector_type(16))) _Float16 v16h;
typedef __attribute__((ext_vector_type(8)))  float    v8f;

#define TSTEPS 512
#define BB     64
#define EE     1024
#define HH     1024

__device__ __forceinline__ v8f wmma_f16(v16h a, v16h b, v8f c) {
  // 8 args: (neg_a, A, neg_b, B, c_mod, C, reuse_a, reuse_b)
  return __builtin_amdgcn_wmma_f32_16x16x32_f16(false, a, false, b,
                                                (short)0, c, false, false);
}

__device__ __forceinline__ float fast_sigmoid(float x) {
  return 1.0f / (1.0f + __expf(-x));
}
__device__ __forceinline__ float fast_tanh(float x) {
  x = fminf(fmaxf(x, -20.0f), 20.0f);
  float e = __expf(2.0f * x);
  return (e - 1.0f) / (e + 1.0f);
}

// Build a 32x16 B-fragment (lane holds K in [khalf*16, khalf*16+16)) from an
// f32 row that is contiguous in K.  p already includes khalf*16.
__device__ __forceinline__ v16h cvt_bfrag_f32(const float* __restrict__ p) {
  float4 f0 = ((const float4*)p)[0];
  float4 f1 = ((const float4*)p)[1];
  float4 f2 = ((const float4*)p)[2];
  float4 f3 = ((const float4*)p)[3];
  union { v16h v; _Float16 h[16]; } u;
  u.h[0]=(_Float16)f0.x; u.h[1]=(_Float16)f0.y; u.h[2]=(_Float16)f0.z; u.h[3]=(_Float16)f0.w;
  u.h[4]=(_Float16)f1.x; u.h[5]=(_Float16)f1.y; u.h[6]=(_Float16)f1.z; u.h[7]=(_Float16)f1.w;
  u.h[8]=(_Float16)f2.x; u.h[9]=(_Float16)f2.y; u.h[10]=(_Float16)f2.z; u.h[11]=(_Float16)f2.w;
  u.h[12]=(_Float16)f3.x; u.h[13]=(_Float16)f3.y; u.h[14]=(_Float16)f3.z; u.h[15]=(_Float16)f3.w;
  return u.v;
}

// ------------------------------ utility kernels ----------------------------

__global__ __launch_bounds__(256) void cvt_f32_to_f16(
    const float* __restrict__ s, _Float16* __restrict__ d, int n4) {
  int i = blockIdx.x * 256 + threadIdx.x;
  if (i < n4) {
    float4 f = ((const float4*)s)[i];
    union { _Float16 h[4]; int2 u; } o;
    o.h[0] = (_Float16)f.x; o.h[1] = (_Float16)f.y;
    o.h[2] = (_Float16)f.z; o.h[3] = (_Float16)f.w;
    ((int2*)d)[i] = o.u;
  }
}

__global__ __launch_bounds__(256) void zero_kernel(unsigned int* p, int n) {
  int i = blockIdx.x * 256 + threadIdx.x;
  if (i < n) p[i] = 0u;
}

// ------------------- phase 1: Zx[g][t] = Wgx @ x_t^T + b_g -----------------
// One wave per (t, bt, ms): 16x16 tile, all 4 gates fused (B-frag reused 4x).
// Output stored in WMMA C-fragment layout: [g][t][ms][bt][lane][8].

__global__ __launch_bounds__(256) void zx_gemm(
    const float*    __restrict__ embeds,   // [T][B][E] f32
    const _Float16* __restrict__ wx4,      // [4H][E] f16
    const float* __restrict__ bgp, const float* __restrict__ bip,
    const float* __restrict__ bfp, const float* __restrict__ bop,
    float* __restrict__ zx) {
  const int tid   = blockIdx.x * 256 + threadIdx.x;
  const int lane  = tid & 31;
  const int w     = tid >> 5;
  const int ms    = w & 63;
  const int bt    = (w >> 6) & 3;
  const int t     = w >> 8;
  const int mrow  = lane & 15;
  const int khalf = lane >> 4;
  const int ncol  = bt * 16 + mrow;
  const int arow  = ms * 16 + mrow;
  const int rbase = ms * 16 + khalf * 8;

  v8f zero;
  #pragma unroll
  for (int r = 0; r < 8; ++r) zero[r] = 0.0f;
  v8f acc[4] = {zero, zero, zero, zero};

  const float*    xrow = embeds + ((size_t)t * BB + ncol) * EE + khalf * 16;
  const _Float16* ab   = wx4 + (size_t)arow * EE + khalf * 16;

  for (int k0 = 0; k0 < EE; k0 += 32) {
    v16h b = cvt_bfrag_f32(xrow + k0);
    #pragma unroll
    for (int g = 0; g < 4; ++g) {
      union { v16h v; int4 q[2]; } a;
      const _Float16* ap = ab + (size_t)g * HH * EE + k0;
      a.q[0] = *(const int4*)(ap);
      a.q[1] = *(const int4*)(ap + 8);
      acc[g] = wmma_f16(a.v, b, acc[g]);
    }
  }

  const float* bptr[4] = {bgp, bip, bfp, bop};
  #pragma unroll
  for (int g = 0; g < 4; ++g) {
    v8f v = acc[g];
    const float* bp = bptr[g] + rbase;
    #pragma unroll
    for (int r = 0; r < 8; ++r) v[r] += bp[r];
    size_t off = (((((size_t)g * TSTEPS + t) * 64 + ms) * 4 + bt) * 32 + lane) * 8;
    *(v8f*)(zx + off) = v;
  }
}

// --------------------------- per-timestep kernel ---------------------------
// 32 blocks x 256 threads = 256 waves; wave (ms, nt) computes rows
// [ms*16, ms*16+16) x cols [nt*16, nt*16+16) of all 4 gate pre-activations,
// then does the cell update entirely in-register.

__global__ __launch_bounds__(256) void lstm_step(
    const _Float16* __restrict__ wh4,    // [4H][H] f16
    const _Float16* __restrict__ wx4,    // [4H][E] f16  (fallback path)
    const float*    __restrict__ x_t,    // embeds + t*B*E (fallback path)
    const float*    __restrict__ zx,     // fragment layout, or nullptr
    const float* __restrict__ bgp, const float* __restrict__ bip,
    const float* __restrict__ bfp, const float* __restrict__ bop,
    const _Float16* __restrict__ h_in,   // [B][H] f16 (h transposed)
    _Float16*       __restrict__ h_out,  // [B][H] f16
    float*          __restrict__ c_st,   // [64][4][32][8] fragment layout
    float*          __restrict__ out_t,  // d_out + t*H*B, [H][B] f32
    int t) {
  const int tid   = blockIdx.x * 256 + threadIdx.x;
  const int lane  = tid & 31;
  const int w     = tid >> 5;
  const int ms    = w & 63;
  const int nt    = (w >> 6) & 3;
  const int mrow  = lane & 15;
  const int khalf = lane >> 4;
  const int arow  = ms * 16 + mrow;
  const int ncol  = nt * 16 + mrow;
  const int rbase = ms * 16 + khalf * 8;

  v8f acc[4];
  if (zx) {
    #pragma unroll
    for (int g = 0; g < 4; ++g) {
      size_t off = (((((size_t)g * TSTEPS + t) * 64 + ms) * 4 + nt) * 32 + lane) * 8;
      acc[g] = *(const v8f*)(zx + off);
    }
  } else {
    const float* bptr[4] = {bgp, bip, bfp, bop};
    #pragma unroll
    for (int g = 0; g < 4; ++g) {
      const float* bp = bptr[g] + rbase;
      v8f a;
      #pragma unroll
      for (int r = 0; r < 8; ++r) a[r] = bp[r];
      acc[g] = a;
    }
    // fused x-projection (only if Zx precompute didn't fit in workspace)
    const float*    xrow = x_t + (size_t)ncol * EE + khalf * 16;
    const _Float16* ab   = wx4 + (size_t)arow * EE + khalf * 16;
    for (int k0 = 0; k0 < EE; k0 += 32) {
      v16h b = cvt_bfrag_f32(xrow + k0);
      #pragma unroll
      for (int g = 0; g < 4; ++g) {
        union { v16h v; int4 q[2]; } a;
        const _Float16* ap = ab + (size_t)g * HH * EE + k0;
        a.q[0] = *(const int4*)(ap);
        a.q[1] = *(const int4*)(ap + 8);
        acc[g] = wmma_f16(a.v, b, acc[g]);
      }
    }
  }

  // recurrent projection: + Wh4 @ h^T
  {
    const _Float16* hrow = h_in + (size_t)ncol * HH + khalf * 16;
    const _Float16* ab   = wh4 + (size_t)arow * HH + khalf * 16;
    for (int k0 = 0; k0 < HH; k0 += 32) {
      union { v16h v; int4 q[2]; } b;
      b.q[0] = *(const int4*)(hrow + k0);
      b.q[1] = *(const int4*)(hrow + k0 + 8);
      #pragma unroll
      for (int g = 0; g < 4; ++g) {
        union { v16h v; int4 q[2]; } a;
        const _Float16* ap = ab + (size_t)g * HH * HH + k0;
        a.q[0] = *(const int4*)(ap);
        a.q[1] = *(const int4*)(ap + 8);
        acc[g] = wmma_f16(a.v, b.v, acc[g]);
      }
    }
  }

  // in-register LSTM cell update (lane owns M = rbase+r, N = ncol)
  const size_t coff = ((size_t)(ms * 4 + nt) * 32 + lane) * 8;
  v8f c = *(const v8f*)(c_st + coff);
  v8f cn, hn;
  #pragma unroll
  for (int r = 0; r < 8; ++r) {
    float gg = fast_tanh(acc[0][r]);
    float ii = fast_sigmoid(acc[1][r]);
    float ff = fast_sigmoid(acc[2][r]);
    float oo = fast_sigmoid(acc[3][r]);
    float cv = gg * ii + c[r] * ff;
    cn[r] = cv;
    hn[r] = fast_tanh(cv) * oo;
  }
  *(v8f*)(c_st + coff) = cn;

  // outputs [H][B] f32
  #pragma unroll
  for (int r = 0; r < 8; ++r)
    out_t[(size_t)(rbase + r) * BB + ncol] = hn[r];

  // h^T f16 for next step: lane's 8 M-rows are contiguous -> one b128 store
  union { int4 q; _Float16 h[8]; } ph;
  #pragma unroll
  for (int r = 0; r < 8; ++r) ph.h[r] = (_Float16)hn[r];
  *(int4*)(h_out + (size_t)ncol * HH + rbase) = ph.q;
}

// ------------------------------- host side ---------------------------------

extern "C" void kernel_launch(void* const* d_in, const int* in_sizes, int n_in,
                              void* d_out, int out_size, void* d_ws, size_t ws_size,
                              hipStream_t stream) {
  (void)in_sizes; (void)n_in; (void)out_size;

  const float* embeds = (const float*)d_in[0];
  // gate order: 0=g(tanh), 1=i, 2=f, 3=o
  const float* Wx[4] = {(const float*)d_in[1], (const float*)d_in[3],
                        (const float*)d_in[5], (const float*)d_in[7]};
  const float* Wh[4] = {(const float*)d_in[2], (const float*)d_in[4],
                        (const float*)d_in[6], (const float*)d_in[8]};
  const float* bgp = (const float*)d_in[9];
  const float* bip = (const float*)d_in[10];
  const float* bfp = (const float*)d_in[11];
  const float* bop = (const float*)d_in[12];

  char* ws = (char*)d_ws;
  _Float16* WX4 = (_Float16*)ws;                       //  8 MB  [4H][E] f16
  _Float16* WH4 = WX4 + (size_t)4 * HH * EE;           //  8 MB  [4H][H] f16
  _Float16* HT0 = WH4 + (size_t)4 * HH * HH;           // 128 KB [B][H] f16
  _Float16* HT1 = HT0 + (size_t)BB * HH;               // 128 KB
  float*    CF  = (float*)(HT1 + (size_t)BB * HH);     // 256 KB c fragments
  float*    ZX  = CF + (size_t)HH * BB;                // 512 MB Zx fragments

  const size_t zx_bytes = (size_t)4 * TSTEPS * HH * BB * sizeof(float);
  const size_t base_bytes = (size_t)((char*)ZX - ws);
  const bool use_zx = ws_size >= base_bytes + zx_bytes;

  // 1) weight conversion to packed f16
  const int n4w = HH * EE / 4;  // == HH*HH/4
  for (int g = 0; g < 4; ++g) {
    cvt_f32_to_f16<<<(n4w + 255) / 256, 256, 0, stream>>>(
        Wx[g], WX4 + (size_t)g * HH * EE, n4w);
    cvt_f32_to_f16<<<(n4w + 255) / 256, 256, 0, stream>>>(
        Wh[g], WH4 + (size_t)g * HH * HH, n4w);
  }

  // 2) zero h0 (both ping-pong buffers) and c state (contiguous region)
  const int zero_dwords = (int)((2 * (size_t)BB * HH * sizeof(_Float16) +
                                 (size_t)HH * BB * sizeof(float)) / 4);
  zero_kernel<<<(zero_dwords + 255) / 256, 256, 0, stream>>>(
      (unsigned int*)HT0, zero_dwords);

  // 3) parallel x-projection GEMM over all timesteps
  if (use_zx) {
    // 512*4*64 waves = 131072 waves = 16384 blocks of 8 waves
    zx_gemm<<<16384, 256, 0, stream>>>(embeds, WX4, bgp, bip, bfp, bop, ZX);
  }

  // 4) sequential recurrence: stream-ordered step kernels
  float* out = (float*)d_out;
  for (int t = 0; t < TSTEPS; ++t) {
    const _Float16* hin  = (t & 1) ? HT1 : HT0;
    _Float16*       hout = (t & 1) ? HT0 : HT1;
    lstm_step<<<32, 256, 0, stream>>>(
        WH4, WX4, embeds + (size_t)t * BB * EE,
        use_zx ? ZX : (const float*)nullptr,
        bgp, bip, bfp, bop,
        hin, hout, CF, out + (size_t)t * HH * BB, t);
  }
}
